// EGAT_78151224918222
// MI455X (gfx1250) — compile-verified, matrix operand order checked by reading, not run on hardware
//
#include <hip/hip_runtime.h>
#include <hip/hip_bf16.h>

// ---------------------------------------------------------------------------
// Types for WMMA fragments (CDNA5 gfx1250, wave32)
// ---------------------------------------------------------------------------
typedef __attribute__((ext_vector_type(16))) __bf16        v16bf;
typedef __attribute__((ext_vector_type(2)))  __bf16        v2bf;
typedef __attribute__((ext_vector_type(8)))  float         v8f;
typedef __attribute__((ext_vector_type(4)))  float         v4f;
typedef __attribute__((ext_vector_type(4)))  unsigned int  v4u;

union Frag { v16bf bf; v4u u[2]; };

#define SLOPE 0.01f

// ---------------------------------------------------------------------------
// Small device helpers
// ---------------------------------------------------------------------------
static __device__ __forceinline__ unsigned f2ord(float f) {
    unsigned u = __float_as_uint(f);
    return (u & 0x80000000u) ? ~u : (u | 0x80000000u);
}
static __device__ __forceinline__ float ord2f(unsigned u) {
    return __uint_as_float((u & 0x80000000u) ? (u & 0x7FFFFFFFu) : ~u);
}
static __device__ __forceinline__ unsigned f2bf_u(float f) {
    unsigned u = __float_as_uint(f);
    return (u + 0x7FFFu + ((u >> 16) & 1u)) >> 16;   // RNE
}
// Pack two f32 into packed bf16x2 (RNE). Prefer hardware v_cvt_pk_bf16_f32.
static __device__ __forceinline__ unsigned pk_bf(float a, float b) {
#if __has_builtin(__builtin_amdgcn_cvt_pk_bf16_f32)
    union { v2bf v; unsigned u; } r;
    r.v = __builtin_amdgcn_cvt_pk_bf16_f32(a, b);
    return r.u;
#else
    return f2bf_u(a) | (f2bf_u(b) << 16);
#endif
}
// Pack 8 contiguous f32 -> 8 bf16 (one v4u), 16B-aligned source
static __device__ __forceinline__ v4u pack8(const float* __restrict__ p) {
    v4f x = *(const v4f*)p;
    v4f y = *(const v4f*)(p + 4);
    v4u r;
    r[0] = pk_bf(x[0], x[1]);
    r[1] = pk_bf(x[2], x[3]);
    r[2] = pk_bf(y[0], y[1]);
    r[3] = pk_bf(y[2], y[3]);
    return r;
}
// Build both K-step A fragments from an f32 row (in-register f32->bf16)
static __device__ __forceinline__ void load_a_f32(const float* __restrict__ arow,
                                                  int hi, Frag& a0, Frag& a1) {
    a0.u[0] = pack8(arow + 8 * hi);
    a0.u[1] = pack8(arow + 16 + 8 * hi);
    a1.u[0] = pack8(arow + 32 + 8 * hi);
    a1.u[1] = pack8(arow + 48 + 8 * hi);
}
// Load both K-step B fragments from a bf16 weight row
static __device__ __forceinline__ void load_b_bf16(const unsigned short* __restrict__ wrow,
                                                   int hi, Frag& b0, Frag& b1) {
    b0.u[0] = *(const v4u*)(wrow + 8 * hi);
    b0.u[1] = *(const v4u*)(wrow + 16 + 8 * hi);
    b1.u[0] = *(const v4u*)(wrow + 32 + 8 * hi);
    b1.u[1] = *(const v4u*)(wrow + 48 + 8 * hi);
}
// 16x16 output tile: 2 WMMAs (K=64), zero-init accumulator (inline SRC2=0)
static __device__ __forceinline__ v8f wmma_tile(const Frag& a0, const Frag& a1,
                                                const Frag& b0, const Frag& b1) {
    v8f c = {};
    c = __builtin_amdgcn_wmma_f32_16x16x32_bf16(false, a0.bf, false, b0.bf,
                                                (short)0, c, false, false);
    c = __builtin_amdgcn_wmma_f32_16x16x32_bf16(false, a1.bf, false, b1.bf,
                                                (short)0, c, false, false);
    return c;
}

// ---------------------------------------------------------------------------
// Elementwise utility kernels
// ---------------------------------------------------------------------------
__global__ void k_cvt_bf16(const float* __restrict__ in,
                           unsigned short* __restrict__ out, int n) {
    int i = blockIdx.x * blockDim.x + threadIdx.x;
    if (i < n) out[i] = (unsigned short)f2bf_u(in[i]);
}

__global__ void k_fill_u32(unsigned int* __restrict__ p, unsigned int v, int n) {
    int i = blockIdx.x * blockDim.x + threadIdx.x;
    if (i < n) p[i] = v;
}

__global__ void k_vec_add(const float* __restrict__ a, const float* __restrict__ b,
                          float* __restrict__ out, int n) {
    int i = blockIdx.x * blockDim.x + threadIdx.x;
    if (i < n) out[i] = a[i] + b[i];
}

// out[n,64] = in[n,0:64] + in[n,64:128]  (sum over H=2 heads)
__global__ void k_head_sum(const float* __restrict__ in, float* __restrict__ out, int total64) {
    int i = blockIdx.x * blockDim.x + threadIdx.x;
    if (i >= total64) return;
    int n = i >> 6, d = i & 63;
    out[i] = in[n * 128 + d] + in[n * 128 + 64 + d];
}

// ---------------------------------------------------------------------------
// Fused node GEMM: loads A[M,64] (f32, converted in-reg to bf16) ONCE and
// computes three projections:
//   Ci = A*Wi^T, Cj = A*Wj^T, Cn = A*Wn^T + bnode          (all [M,128] f32)
// One wave -> 16 rows; 48 WMMAs per wave. Bias is added at store time so all
// WMMAs use the inline-0 C operand (no accumulator broadcast movs).
// ---------------------------------------------------------------------------
__global__ __launch_bounds__(256) void k_gemm3_node(
    const float* __restrict__ A,
    const unsigned short* __restrict__ Wi,
    const unsigned short* __restrict__ Wj,
    const unsigned short* __restrict__ Wn,
    const float* __restrict__ bnode,
    float* __restrict__ Ci, float* __restrict__ Cj, float* __restrict__ Cn,
    int M)
{
    const int wave = threadIdx.x >> 5;
    const int lane = threadIdx.x & 31;
    const int rowBase = (blockIdx.x * 8 + wave) * 16;
    if (rowBase >= M) return;               // uniform across the wave

    const int mr = lane & 15;
    const int hi = lane >> 4;

    Frag a0, a1;
    load_a_f32(A + (rowBase + mr) * 64, hi, a0, a1);

    const unsigned short* Ws[3] = { Wi, Wj, Wn };
    float*               Cs[3] = { Ci, Cj, Cn };

#pragma unroll
    for (int w = 0; w < 3; ++w) {
        const unsigned short* W = Ws[w];
        float* C = Cs[w];
#pragma unroll
        for (int tn = 0; tn < 8; ++tn) {
            const int n = tn * 16 + mr;
            Frag b0, b1;
            load_b_bf16(W + n * 64, hi, b0, b1);

            v8f c = wmma_tile(a0, a1, b0, b1);
            const float bv = (w == 2) ? bnode[n] : 0.0f;
#pragma unroll
            for (int v = 0; v < 8; ++v)
                C[(rowBase + 8 * hi + v) * 128 + n] = c[v] + bv;
        }
    }
}

// ---------------------------------------------------------------------------
// Edge GEMM: C[M,128] = A[M,64] (f32 -> bf16 in-reg) * W[128,64]^T, no bias.
// ---------------------------------------------------------------------------
__global__ __launch_bounds__(256) void k_gemm_edge(
    const float* __restrict__ A,
    const unsigned short* __restrict__ W,
    float* __restrict__ C,
    int M)
{
    const int wave = threadIdx.x >> 5;
    const int lane = threadIdx.x & 31;
    const int rowBase = (blockIdx.x * 8 + wave) * 16;
    if (rowBase >= M) return;

    const int mr = lane & 15;
    const int hi = lane >> 4;

    Frag a0, a1;
    load_a_f32(A + (rowBase + mr) * 64, hi, a0, a1);

#pragma unroll
    for (int tn = 0; tn < 8; ++tn) {
        const int n = tn * 16 + mr;
        Frag b0, b1;
        load_b_bf16(W + n * 64, hi, b0, b1);

        v8f c = wmma_tile(a0, a1, b0, b1);
#pragma unroll
        for (int v = 0; v < 8; ++v)
            C[(rowBase + 8 * hi + v) * 128 + n] = c[v];
    }
}

// ---------------------------------------------------------------------------
// Edge kernel 1 (wave per edge):
//   fo = f_ni[src] + f_nj[dst] + f_fij[e] + bias ; leaky_relu ; store f_out
//   logit[e,h] = dot(fo[h*64:(h+1)*64], attn[h])  ; atomic segment-max on dst
// ---------------------------------------------------------------------------
__global__ __launch_bounds__(256) void k_edge_attn(
    const int* __restrict__ src, const int* __restrict__ dst,
    const float* __restrict__ fni, const float* __restrict__ fnj,
    const float* __restrict__ ffij, const float* __restrict__ bias,
    const float* __restrict__ attn,   // [128] flat [H,D]
    float* __restrict__ fout,         // [E,128] (may alias ffij)
    float* __restrict__ logit,        // [E,2]
    unsigned int* __restrict__ emax_ord,  // [N,2], pre-filled with f2ord(-inf)
    int E)
{
    const int wave = threadIdx.x >> 5, lane = threadIdx.x & 31;
    const int e = blockIdx.x * 8 + wave;
    if (e >= E) return;
    const int s = src[e], d = dst[e];
    const float* pi = fni + s * 128;
    const float* pj = fnj + d * 128;
    const float* pf = ffij + e * 128;
    float p0 = 0.f, p1 = 0.f;
#pragma unroll
    for (int t = 0; t < 4; ++t) {
        const int j = lane + t * 32;
        float v = pi[j] + pj[j] + pf[j] + bias[j];
        v = v > 0.f ? v : v * SLOPE;
        fout[e * 128 + j] = v;
        const float c = v * attn[j];
        if (t < 2) p0 += c; else p1 += c;
    }
#pragma unroll
    for (int off = 16; off; off >>= 1) {
        p0 += __shfl_xor(p0, off, 32);
        p1 += __shfl_xor(p1, off, 32);
    }
    if (lane == 0) {
        logit[e * 2 + 0] = p0;
        logit[e * 2 + 1] = p1;
        atomicMax(&emax_ord[d * 2 + 0], f2ord(p0));
        atomicMax(&emax_ord[d * 2 + 1], f2ord(p1));
    }
}

// ee = exp(logit - max[dst]);  den[dst] += ee
__global__ void k_edge_exp(const int* __restrict__ dst,
                           const float* __restrict__ logit,
                           const unsigned int* __restrict__ emax_ord,
                           float* __restrict__ ee, float* __restrict__ den, int E) {
    int e = blockIdx.x * blockDim.x + threadIdx.x;
    if (e >= E) return;
    int d = dst[e];
#pragma unroll
    for (int h = 0; h < 2; ++h) {
        float m = ord2f(emax_ord[d * 2 + h]);
        float v = __expf(logit[e * 2 + h] - m);
        ee[e * 2 + h] = v;
        unsafeAtomicAdd(&den[d * 2 + h], v);
    }
}

// h_out[dst] += h[src] * a   (wave per edge, 128 features)
__global__ __launch_bounds__(256) void k_edge_scatter(
    const int* __restrict__ src, const int* __restrict__ dst,
    const float* __restrict__ ee, const float* __restrict__ den,
    const float* __restrict__ hm, float* __restrict__ hout, int E)
{
    const int wave = threadIdx.x >> 5, lane = threadIdx.x & 31;
    const int e = blockIdx.x * 8 + wave;
    if (e >= E) return;
    const int s = src[e], d = dst[e];
    const float a0 = ee[e * 2 + 0] / den[d * 2 + 0];
    const float a1 = ee[e * 2 + 1] / den[d * 2 + 1];
    const float* ph = hm + s * 128;
    float* po = hout + d * 128;
#pragma unroll
    for (int t = 0; t < 4; ++t) {
        const int j = lane + t * 32;
        unsafeAtomicAdd(&po[j], ph[j] * ((t < 2) ? a0 : a1));
    }
}

// Column sums over [N,64] into out[64] (out must be zeroed first)
__global__ __launch_bounds__(256) void k_col_sum(const float* __restrict__ in,
                                                 float* __restrict__ out, int N) {
    __shared__ float sm[256];
    const int d = threadIdx.x & 63, g = threadIdx.x >> 6;
    float acc = 0.f;
    for (int n = blockIdx.x * 4 + g; n < N; n += gridDim.x * 4)
        acc += in[n * 64 + d];
    sm[threadIdx.x] = acc;
    __syncthreads();
    if (g == 0)
        unsafeAtomicAdd(&out[d], sm[d] + sm[64 + d] + sm[128 + d] + sm[192 + d]);
}

// Readout MLP: y = (y1sum+inisum)*invN ; hdn = sig(y@W1^T+b1) ; out = sig(hdn@W2^T+b2)
__global__ __launch_bounds__(128) void k_readout(
    const float* __restrict__ y1sum, const float* __restrict__ inisum,
    const float* __restrict__ W1, const float* __restrict__ b1,
    const float* __restrict__ W2, const float* __restrict__ b2,
    float* __restrict__ out, float invN)
{
    __shared__ float y[64];
    __shared__ float hdn[128];
    const int t = threadIdx.x;
    if (t < 64) y[t] = (y1sum[t] + inisum[t]) * invN;
    __syncthreads();
    float acc = b1[t];
#pragma unroll
    for (int k = 0; k < 64; ++k) acc += y[k] * W1[t * 64 + k];
    hdn[t] = 1.f / (1.f + __expf(-acc));
    __syncthreads();
    if (t < 16) {
        float a = b2[t];
#pragma unroll
        for (int k = 0; k < 128; ++k) a += hdn[k] * W2[t * 128 + k];
        out[t] = 1.f / (1.f + __expf(-a));
    }
}

// ---------------------------------------------------------------------------
// Host orchestration
// ---------------------------------------------------------------------------
extern "C" void kernel_launch(void* const* d_in, const int* in_sizes, int n_in,
                              void* d_out, int out_size, void* d_ws, size_t ws_size,
                              hipStream_t stream) {
    (void)in_sizes; (void)n_in; (void)out_size; (void)ws_size;

    const int NGG = 400000, EGG = 800000, NLG = 50000, ELG = 400000;

    const float* gg_nf = (const float*)d_in[0];
    const float* gg_ef = (const float*)d_in[1];
    const float* lg_nf = (const float*)d_in[2];
    const float* lg_ef = (const float*)d_in[3];
    const float* ro_W1 = (const float*)d_in[25];
    const float* ro_b1 = (const float*)d_in[26];
    const float* ro_W2 = (const float*)d_in[27];
    const float* ro_b2 = (const float*)d_in[28];
    const int* gg_src = (const int*)d_in[29];
    const int* gg_dst = (const int*)d_in[30];
    const int* lg_src = (const int*)d_in[31];
    const int* lg_dst = (const int*)d_in[32];

    // -------- workspace bump allocator (256B aligned) --------
    char* base = (char*)d_ws;
    size_t off = 0;
    auto alloc = [&](size_t bytes) -> void* {
        void* p = base + off;
        off += (bytes + 255) & ~(size_t)255;
        return p;
    };
    unsigned short* wn_bf = (unsigned short*)alloc(128 * 64 * 2);
    unsigned short* wi_bf = (unsigned short*)alloc(128 * 64 * 2);
    unsigned short* wj_bf = (unsigned short*)alloc(128 * 64 * 2);
    unsigned short* wf_bf = (unsigned short*)alloc(128 * 64 * 2);
    float* fni   = (float*)alloc((size_t)NGG * 128 * 4);
    float* fnj   = (float*)alloc((size_t)NGG * 128 * 4);
    float* hm    = (float*)alloc((size_t)NGG * 128 * 4);
    float* ffij  = (float*)alloc((size_t)EGG * 128 * 4);   // also holds f_out
    float* hout  = (float*)alloc((size_t)NGG * 128 * 4);
    float* logit = (float*)alloc((size_t)EGG * 2 * 4);
    float* ee    = (float*)alloc((size_t)EGG * 2 * 4);
    unsigned int* emax = (unsigned int*)alloc((size_t)NGG * 2 * 4);
    float* den   = (float*)alloc((size_t)NGG * 2 * 4);
    float* y2    = (float*)alloc((size_t)NGG * 64 * 4);    // gg node out (== lg edge count)
    float* nfe   = (float*)alloc((size_t)NLG * 64 * 4);
    float* nfe2  = (float*)alloc((size_t)NLG * 64 * 4);
    float* efe   = (float*)alloc((size_t)ELG * 64 * 4);
    float* y1sum = (float*)alloc(64 * 4);
    float* inisum= (float*)alloc(64 * 4);

    auto cdiv = [](int a, int b) { return (a + b - 1) / b; };

    // One EGAT layer. node_out: [N,64]; edge_out (may be null): [E,64]
    auto run_layer = [&](const int* src, const int* dst,
                         const float* nf, int N, const float* ef, int E,
                         int li, float* node_out, float* edge_out) {
        const float* Wnode = (const float*)d_in[4 + li * 7 + 0];
        const float* bnode = (const float*)d_in[4 + li * 7 + 1];
        const float* Wni   = (const float*)d_in[4 + li * 7 + 2];
        const float* Wnj   = (const float*)d_in[4 + li * 7 + 3];
        const float* Wfij  = (const float*)d_in[4 + li * 7 + 4];
        const float* attn  = (const float*)d_in[4 + li * 7 + 5];
        const float* bias  = (const float*)d_in[4 + li * 7 + 6];

        // bf16 conversion for the (tiny) weight matrices only
        k_cvt_bf16<<<cdiv(128 * 64, 256), 256, 0, stream>>>(Wnode, wn_bf, 128 * 64);
        k_cvt_bf16<<<cdiv(128 * 64, 256), 256, 0, stream>>>(Wni,   wi_bf, 128 * 64);
        k_cvt_bf16<<<cdiv(128 * 64, 256), 256, 0, stream>>>(Wnj,   wj_bf, 128 * 64);
        k_cvt_bf16<<<cdiv(128 * 64, 256), 256, 0, stream>>>(Wfij,  wf_bf, 128 * 64);

        // WMMA GEMMs: node projections fused (A read once), edge projection
        k_gemm3_node<<<cdiv(N, 128), 256, 0, stream>>>(nf, wi_bf, wj_bf, wn_bf, bnode,
                                                       fni, fnj, hm, N);
        k_gemm_edge<<<cdiv(E, 128), 256, 0, stream>>>(ef, wf_bf, ffij, E);

        // init softmax / accumulation state
        k_fill_u32<<<cdiv(N * 2, 256), 256, 0, stream>>>(emax, 0x007FFFFFu, N * 2); // ord(-inf)
        k_fill_u32<<<cdiv(N * 2, 256), 256, 0, stream>>>((unsigned*)den, 0u, N * 2);
        k_fill_u32<<<cdiv(N * 128, 256), 256, 0, stream>>>((unsigned*)hout, 0u, N * 128);

        // edge softmax + aggregation
        k_edge_attn<<<cdiv(E, 8), 256, 0, stream>>>(src, dst, fni, fnj, ffij, bias,
                                                    attn, ffij, logit, emax, E);
        k_edge_exp<<<cdiv(E, 256), 256, 0, stream>>>(dst, logit, emax, ee, den, E);
        k_edge_scatter<<<cdiv(E, 8), 256, 0, stream>>>(src, dst, ee, den, hm, hout, E);

        // head sums
        k_head_sum<<<cdiv(N * 64, 256), 256, 0, stream>>>(hout, node_out, N * 64);
        if (edge_out)
            k_head_sum<<<cdiv(E * 64, 256), 256, 0, stream>>>(ffij, edge_out, E * 64);
    };

    // ---- gcn_1 on gg ----
    run_layer(gg_src, gg_dst, gg_nf, NGG, gg_ef, EGG, /*l1*/0, y2, nullptr);

    // efe = y2 + lg_edge_feat ; ini = mean(lg_node_feat)
    k_vec_add<<<cdiv(ELG * 64, 256), 256, 0, stream>>>(y2, lg_ef, efe, ELG * 64);
    k_fill_u32<<<1, 64, 0, stream>>>((unsigned*)inisum, 0u, 64);
    k_col_sum<<<256, 256, 0, stream>>>(lg_nf, inisum, NLG);

    // ---- gcn_2 (two layers) on lg ----
    run_layer(lg_src, lg_dst, lg_nf, NLG, efe, ELG, /*l2a*/1, nfe, efe);
    run_layer(lg_src, lg_dst, nfe,   NLG, efe, ELG, /*l2b*/2, nfe2, nullptr);

    // y1 = mean(nfe2) ; readout MLP
    k_fill_u32<<<1, 64, 0, stream>>>((unsigned*)y1sum, 0u, 64);
    k_col_sum<<<256, 256, 0, stream>>>(nfe2, y1sum, NLG);
    k_readout<<<1, 128, 0, stream>>>(y1sum, inisum, ro_W1, ro_b1, ro_W2, ro_b2,
                                     (float*)d_out, 1.0f / (float)NLG);
}
